// WindowCrossAttention_34170759807004
// MI455X (gfx1250) — matrix operand order repaired and checked by reading, not compiled, not run on hardware
//
#include <hip/hip_runtime.h>
#include <hip/hip_bf16.h>

typedef __attribute__((ext_vector_type(16))) _Float16 v16h;
typedef __attribute__((ext_vector_type(8)))  float    v8f;

#define DIM   192
#define HEADS 6
#define HD    32
#define NTOK  64
#define NW    1024

// ---------------- WMMA fragment helpers (layouts per CDNA5 ISA §7.12.2) ----

// A 16x32 f16 frag from f32 global matrix (row-major, stride ld), convert on the fly.
__device__ inline v16h frag_a_f32(const float* __restrict__ X, int ld, int row0, int k0, int lane) {
  const float* p = X + (size_t)(row0 + (lane & 15)) * ld + k0 + ((lane >> 4) << 3);
  v16h a;
#pragma unroll
  for (int i = 0; i < 8; ++i) { a[i] = (_Float16)p[i]; a[8 + i] = (_Float16)p[16 + i]; }
  return a;
}

// A 16x32 f16 frag from f16 matrix (row-major, stride ld).
__device__ inline v16h frag_a_h(const _Float16* X, int ld, int row0, int k0, int lane) {
  const _Float16* p = X + (size_t)(row0 + (lane & 15)) * ld + k0 + ((lane >> 4) << 3);
  v16h a;
#pragma unroll
  for (int i = 0; i < 8; ++i) { a[i] = p[i]; a[8 + i] = p[16 + i]; }
  return a;
}

// B 32x16 frag where B[k][n] == W[n][k] (row-major, stride ld): per-lane k = 8h+i and 16+8h+i.
__device__ inline v16h frag_b_nmaj(const _Float16* W, int ld, int k0, int n0, int lane) {
  const _Float16* p = W + (size_t)(n0 + (lane & 15)) * ld + k0 + ((lane >> 4) << 3);
  v16h b;
#pragma unroll
  for (int i = 0; i < 8; ++i) { b[i] = p[i]; b[8 + i] = p[16 + i]; }
  return b;
}

__device__ inline v8f wmma_f16(v16h a, v16h b, v8f c) {
  return __builtin_amdgcn_wmma_f32_16x16x32_f16(false, a, false, b, (short)0, c, false, false);
}

// ---------------- prologue: f32 -> f16 weight conversion, rel-bias gather -------------

__global__ void cvt_f16(const float* __restrict__ s, _Float16* __restrict__ d, int n) {
  int i = blockIdx.x * 256 + threadIdx.x;
  if (i < n) d[i] = (_Float16)s[i];
}

__global__ void expand_bias(const float* __restrict__ tbl, const int* __restrict__ rel,
                            float* __restrict__ rb) {
  int i = blockIdx.x * 256 + threadIdx.x;
  if (i < HEADS * NTOK * NTOK) {
    int h = i >> 12;          // i / 4096
    int nm = i & 4095;        // n*64 + m
    rb[i] = tbl[rel[nm] * HEADS + h];
  }
}

// ---------------- fused window cross-attention: one block per window ------------------

__global__ __launch_bounds__(256) void fused_wca(
    const float* __restrict__ x, const float* __restrict__ y,
    const float* __restrict__ mask_x, const float* __restrict__ mask_y,
    const float* __restrict__ qkv_b, const float* __restrict__ kv_b,
    const float* __restrict__ m1b, const float* __restrict__ m2b,
    const float* __restrict__ pjb,
    const _Float16* __restrict__ qkv_w, const _Float16* __restrict__ kv_w,
    const _Float16* __restrict__ m1w, const _Float16* __restrict__ m2w,
    const _Float16* __restrict__ pjw,
    const float* __restrict__ rbx, const float* __restrict__ rby,
    float* __restrict__ out)
{
  __shared__ _Float16 sQ[NTOK * DIM];          // 24 KB  q (pre-scaled), [n][c]
  __shared__ _Float16 sK[NTOK * DIM];          // 24 KB  k / ky, [n][c]; later merge2 out
  __shared__ _Float16 sVt[DIM * NTOK];         // 24 KB  v / vy TRANSPOSED: [c][n]
  __shared__ unsigned char sSP[24576];         // 24 KB  S f32 (16K) + P f16 (8K); later merge1 out
  __shared__ _Float16 sXO[NTOK * 2 * DIM];     // 48 KB  concat [x_out | y_out], f16

  float*    S  = (float*)sSP;
  _Float16* P  = (_Float16*)(sSP + 16384);
  _Float16* Hb = (_Float16*)sSP;               // aliases S/P after attention
  _Float16* M2 = sK;                           // aliases ky after attention

  const int b    = blockIdx.x;
  const int tid  = threadIdx.x;
  const int lane = tid & 31;
  const int wv   = tid >> 5;                   // 8 waves
  const int wi   = b & (NW - 1);               // window-mask index: b % nW
  const float scale = 0.17677669529663689f;    // 1/sqrt(32)

  const float* xb = x + (size_t)b * NTOK * DIM;
  const float* yb = y + (size_t)b * NTOK * DIM;

  // ---- Stage 1: qkv = x @ qkv_w.T + qkv_b -> sQ (scaled), sK, sVt  (M64 x N576, K192)
  // Wave owns one M-tile; A fragments register-blocked across all its N-tiles.
  {
    const int mt = wv & 3;
    const int r0 = mt * 16 + ((lane >> 4) << 3);
    v16h af[6];
#pragma unroll
    for (int kt = 0; kt < 6; ++kt) af[kt] = frag_a_f32(xb, DIM, mt * 16, kt * 32, lane);
    const int nt0 = (wv >> 2) * 18;
    for (int nt = nt0; nt < nt0 + 18; ++nt) {
      v8f acc = {};
#pragma unroll
      for (int kt = 0; kt < 6; ++kt)
        acc = wmma_f16(af[kt], frag_b_nmaj(qkv_w, DIM, kt * 32, nt * 16, lane), acc);
      int o = nt * 16 + (lane & 15);
      float bias = qkv_b[o];
      if (nt < 12) {                                  // q: scale + store [n][c]
#pragma unroll
        for (int r = 0; r < 8; ++r) sQ[(r0 + r) * DIM + o] = (_Float16)((acc[r] + bias) * scale);
      } else if (nt < 24) {                           // k: store [n][c]
        int c = o - DIM;
#pragma unroll
        for (int r = 0; r < 8; ++r) sK[(r0 + r) * DIM + c] = (_Float16)(acc[r] + bias);
      } else {                                        // v: store transposed [c][n]
        int c = o - 2 * DIM;
#pragma unroll
        for (int r = 0; r < 8; ++r) sVt[c * NTOK + (r0 + r)] = (_Float16)(acc[r] + bias);
      }
    }
  }
  __syncthreads();

  // ---- Stage 2: two attention phases (src 0: x with k,v ; src 1: y with ky,vy) ------
  for (int src = 0; src < 2; ++src) {
    if (src == 1) {
      // kv = y @ kv_w.T + kv_b -> sK, sVt   (M64 x N384, K192)
      const int mt = wv & 3;
      const int r0 = mt * 16 + ((lane >> 4) << 3);
      v16h af[6];
#pragma unroll
      for (int kt = 0; kt < 6; ++kt) af[kt] = frag_a_f32(yb, DIM, mt * 16, kt * 32, lane);
      const int nt0 = (wv >> 2) * 12;
      for (int nt = nt0; nt < nt0 + 12; ++nt) {
        v8f acc = {};
#pragma unroll
        for (int kt = 0; kt < 6; ++kt)
          acc = wmma_f16(af[kt], frag_b_nmaj(kv_w, DIM, kt * 32, nt * 16, lane), acc);
        int o = nt * 16 + (lane & 15);
        float bias = kv_b[o];
        if (nt < 12) {
#pragma unroll
          for (int r = 0; r < 8; ++r) sK[(r0 + r) * DIM + o] = (_Float16)(acc[r] + bias);
        } else {
          int c = o - DIM;
#pragma unroll
          for (int r = 0; r < 8; ++r) sVt[c * NTOK + (r0 + r)] = (_Float16)(acc[r] + bias);
        }
      }
      __syncthreads();
    }

    const float* rb = src ? rby : rbx;
    const float* mk = (src ? mask_y : mask_x) + (size_t)wi * NTOK * NTOK;

    for (int h = 0; h < HEADS; ++h) {
      // S = q_h @ k_h^T (raw scores)   (64x64, K=32 -> 1 WMMA per tile, 16 tiles)
      // Each wave: one M-tile (A frag loaded once), two N-tiles.
      {
        int mt = wv >> 1;
        v16h a = frag_a_h(sQ, DIM, mt * 16, h * HD, lane);
        int r0 = mt * 16 + ((lane >> 4) << 3);
#pragma unroll
        for (int j = 0; j < 2; ++j) {
          int nt = (wv & 1) * 2 + j;
          v16h bf = frag_b_nmaj(sK + h * HD, DIM, 0, nt * 16, lane);  // B[d][m] = k[m][d]
          v8f acc = {};
          acc = wmma_f16(a, bf, acc);
          int col = nt * 16 + (lane & 15);
#pragma unroll
          for (int r = 0; r < 8; ++r) S[(r0 + r) * NTOK + col] = acc[r];
        }
      }
      __syncthreads();

      // softmax over 64 cols (adds rel_bias + mask here: contiguous coalesced reads);
      // 4 lanes per row, quad reduction via shfl_xor
      {
        const float* rbh = rb + h * NTOK * NTOK;
        int row = tid >> 2;
        int c0  = (tid & 3) * 16;
        float sv[16];
#pragma unroll
        for (int i = 0; i < 16; ++i)
          sv[i] = S[row * NTOK + c0 + i] + rbh[row * NTOK + c0 + i] + mk[row * NTOK + c0 + i];
        float mx = -3.4e38f;
#pragma unroll
        for (int i = 0; i < 16; ++i) mx = fmaxf(mx, sv[i]);
        mx = fmaxf(mx, __shfl_xor(mx, 1));
        mx = fmaxf(mx, __shfl_xor(mx, 2));
        float sum = 0.f;
#pragma unroll
        for (int i = 0; i < 16; ++i) { sv[i] = __expf(sv[i] - mx); sum += sv[i]; }
        sum += __shfl_xor(sum, 1);
        sum += __shfl_xor(sum, 2);
        float inv = 1.f / sum;
#pragma unroll
        for (int i = 0; i < 16; ++i) P[row * NTOK + c0 + i] = (_Float16)(sv[i] * inv);
      }
      __syncthreads();

      // O = P @ v_h  (64x32, K=64 -> 2 WMMA per tile, 8 tiles = 1 per wave)
      // B[k=m][n=d] = v[m][d] = sVt[d][m]  -> n-major contiguous loads
      {
        int mt = wv >> 1, nt = wv & 1;
        v8f acc = {};
#pragma unroll
        for (int kt = 0; kt < 2; ++kt) {
          v16h a  = frag_a_h(P, NTOK, mt * 16, kt * 32, lane);
          v16h bf = frag_b_nmaj(sVt + (h * HD) * NTOK, NTOK, kt * 32, nt * 16, lane);
          acc = wmma_f16(a, bf, acc);
        }
        int col = src * DIM + h * HD + nt * 16 + (lane & 15);
        int r0  = mt * 16 + ((lane >> 4) << 3);
#pragma unroll
        for (int r = 0; r < 8; ++r) sXO[(r0 + r) * (2 * DIM) + col] = (_Float16)acc[r];
      }
      __syncthreads();
    }
  }

  // ---- Stage 3: merge1 = LeakyReLU(concat @ m1w.T + m1b)  (M64 x N192, K384) --------
  for (int t = wv; t < 48; t += 8) {
    int mt = t / 12, nt = t % 12;
    v8f acc = {};
#pragma unroll
    for (int kt = 0; kt < 12; ++kt) {
      v16h a  = frag_a_h(sXO, 2 * DIM, mt * 16, kt * 32, lane);
      v16h bf = frag_b_nmaj(m1w, 2 * DIM, kt * 32, nt * 16, lane);
      acc = wmma_f16(a, bf, acc);
    }
    int col = nt * 16 + (lane & 15);
    float bias = m1b[col];
    int r0 = mt * 16 + ((lane >> 4) << 3);
#pragma unroll
    for (int r = 0; r < 8; ++r) {
      float vv = acc[r] + bias;
      vv = vv >= 0.f ? vv : 0.2f * vv;
      Hb[(r0 + r) * DIM + col] = (_Float16)vv;
    }
  }
  __syncthreads();

  // ---- Stage 4: merge2 = Hb @ m2w.T + m2b + x_out  (residual) -----------------------
  for (int t = wv; t < 48; t += 8) {
    int mt = t / 12, nt = t % 12;
    v8f acc = {};
#pragma unroll
    for (int kt = 0; kt < 6; ++kt) {
      v16h a  = frag_a_h(Hb, DIM, mt * 16, kt * 32, lane);
      v16h bf = frag_b_nmaj(m2w, DIM, kt * 32, nt * 16, lane);
      acc = wmma_f16(a, bf, acc);
    }
    int col = nt * 16 + (lane & 15);
    float bias = m2b[col];
    int r0 = mt * 16 + ((lane >> 4) << 3);
#pragma unroll
    for (int r = 0; r < 8; ++r) {
      int row = r0 + r;
      float vv = acc[r] + bias + (float)sXO[row * (2 * DIM) + col];  // x_out residual
      M2[row * DIM + col] = (_Float16)vv;
    }
  }
  __syncthreads();

  // ---- Stage 5: out = M2 @ proj_w.T + proj_b -> global f32 --------------------------
  float* ob = out + (size_t)b * NTOK * DIM;
  for (int t = wv; t < 48; t += 8) {
    int mt = t / 12, nt = t % 12;
    v8f acc = {};
#pragma unroll
    for (int kt = 0; kt < 6; ++kt) {
      v16h a  = frag_a_h(M2, DIM, mt * 16, kt * 32, lane);
      v16h bf = frag_b_nmaj(pjw, DIM, kt * 32, nt * 16, lane);
      acc = wmma_f16(a, bf, acc);
    }
    int col = nt * 16 + (lane & 15);
    float bias = pjb[col];
    int r0 = mt * 16 + ((lane >> 4) << 3);
#pragma unroll
    for (int r = 0; r < 8; ++r) ob[(r0 + r) * DIM + col] = acc[r] + bias;
  }
}

// ---------------- host glue -----------------------------------------------------------

extern "C" void kernel_launch(void* const* d_in, const int* in_sizes, int n_in,
                              void* d_out, int out_size, void* d_ws, size_t ws_size,
                              hipStream_t stream) {
  (void)in_sizes; (void)n_in; (void)out_size; (void)ws_size;
  const float* x      = (const float*)d_in[0];
  const float* y      = (const float*)d_in[1];
  const float* mask_x = (const float*)d_in[2];
  const float* mask_y = (const float*)d_in[3];
  const float* qkv_w  = (const float*)d_in[4];
  const float* qkv_b  = (const float*)d_in[5];
  const float* kv_w   = (const float*)d_in[6];
  const float* kv_b   = (const float*)d_in[7];
  const float* m1w    = (const float*)d_in[8];
  const float* m1b    = (const float*)d_in[9];
  const float* m2w    = (const float*)d_in[10];
  const float* m2b    = (const float*)d_in[11];
  const float* pjw    = (const float*)d_in[12];
  const float* pjb    = (const float*)d_in[13];
  const float* btx    = (const float*)d_in[14];
  const float* bty    = (const float*)d_in[15];
  const int*   rel    = (const int*)d_in[16];

  char* ws = (char*)d_ws;
  _Float16* qkv_wh = (_Float16*)(ws + 0);       // 576*192 h = 221184 B
  _Float16* kv_wh  = (_Float16*)(ws + 221184);  // 384*192 h = 147456 B
  _Float16* m1wh   = (_Float16*)(ws + 368640);  // 192*384 h = 147456 B
  _Float16* m2wh   = (_Float16*)(ws + 516096);  // 192*192 h =  73728 B
  _Float16* pjwh   = (_Float16*)(ws + 589824);  // 192*192 h =  73728 B
  float*    rbx    = (float*)(ws + 663552);     // 6*64*64 f =  98304 B
  float*    rby    = (float*)(ws + 761856);     // 6*64*64 f =  98304 B

  cvt_f16<<<(110592 + 255) / 256, 256, 0, stream>>>(qkv_w, qkv_wh, 110592);
  cvt_f16<<<(73728 + 255) / 256, 256, 0, stream>>>(kv_w, kv_wh, 73728);
  cvt_f16<<<(73728 + 255) / 256, 256, 0, stream>>>(m1w, m1wh, 73728);
  cvt_f16<<<(36864 + 255) / 256, 256, 0, stream>>>(m2w, m2wh, 36864);
  cvt_f16<<<(36864 + 255) / 256, 256, 0, stream>>>(pjw, pjwh, 36864);
  expand_bias<<<(24576 + 255) / 256, 256, 0, stream>>>(btx, rel, rbx);
  expand_bias<<<(24576 + 255) / 256, 256, 0, stream>>>(bty, rel, rby);

  fused_wca<<<8192, 256, 0, stream>>>(x, y, mask_x, mask_y, qkv_b, kv_b, m1b, m2b, pjb,
                                      qkv_wh, kv_wh, m1wh, m2wh, pjwh, rbx, rby,
                                      (float*)d_out);
}
